// SLRU_89928025244451
// MI455X (gfx1250) — compile-verified, matrix operand order checked by reading, not run on hardware
//
#include <hip/hip_runtime.h>
#include <stdint.h>

typedef __bf16 bf16;
typedef float v8f  __attribute__((ext_vector_type(8)));
typedef bf16  v16bf __attribute__((ext_vector_type(16)));
typedef int   v4i_gcc __attribute__((vector_size(4 * sizeof(int))));  // matches builtin param

struct alignas(16) U4 { unsigned x, y, z, w; };
union Frag { v16bf v; U4 q[2]; };

#if defined(__has_builtin)
#  if __has_builtin(__builtin_amdgcn_global_load_async_to_lds_b128)
#    define HAVE_ASYNC_LDS 1
#  endif
#  if __has_builtin(__builtin_amdgcn_s_wait_asynccnt)
#    define HAVE_WAIT_ASYNC_BUILTIN 1
#  endif
#endif

#define AS1q __attribute__((address_space(1)))
#define AS3q __attribute__((address_space(3)))

// CK-style: LDS generic pointer -> addrspace(3) via integer truncation (low 32 bits = LDS offset)
__device__ __forceinline__ AS3q v4i_gcc* to_lds_v4(void* p) {
  return (AS3q v4i_gcc*)(uintptr_t)p;
}
__device__ __forceinline__ AS1q v4i_gcc* to_global_v4(const void* p) {
  return (AS1q v4i_gcc*)(uintptr_t)p;
}

__device__ __forceinline__ void wait_async_le8() {
#if HAVE_ASYNC_LDS
#  if HAVE_WAIT_ASYNC_BUILTIN
  __builtin_amdgcn_s_wait_asynccnt(8);
#  else
  asm volatile("s_wait_asynccnt 8" ::: "memory");
#  endif
#endif
}
__device__ __forceinline__ void wait_async_le0() {
#if HAVE_ASYNC_LDS
#  if HAVE_WAIT_ASYNC_BUILTIN
  __builtin_amdgcn_s_wait_asynccnt(0);
#  else
  asm volatile("s_wait_asynccnt 0" ::: "memory");
#  endif
#endif
}

#define B_      4
#define L_      4096
#define H_      1024
#define U_      1024
#define BL_     16384
#define NCHUNK  64
#define CHUNK   64

#define TM 128
#define TN 128
#define TK 64
#define PITCH 144  // 64 bf16 = 128B data + 16B pad: 16B-aligned, conflict-free ds_load_b128

// ---------------------------------------------------------------- convert
__global__ __launch_bounds__(256) void cvt_f32_bf16(const float* __restrict__ in,
                                                    bf16* __restrict__ out, int n) {
  int i = blockIdx.x * 256 + threadIdx.x;
  if (i < n) out[i] = (bf16)in[i];
}

// ---------------------------------------------------------------- GEMM: out = A(MxK) * W(NxK)^T + bias
// Double-buffered: async-copy 128x64 tiles of A and W for step k+1 into LDS
// while 16 WMMAs consume step k.
template <bool OUT_F32>
__global__ __launch_bounds__(256) void gemm_bf16_wmma(
    const bf16* __restrict__ A, const bf16* __restrict__ W,
    const float* __restrict__ bias, float* __restrict__ outF,
    bf16* __restrict__ outB, int M, int N, int K)
{
  __shared__ unsigned char sA[2][TM * PITCH];
  __shared__ unsigned char sB[2][TN * PITCH];

  const int t    = threadIdx.x;
  const int lane = t & 31;
  const int w    = t >> 5;     // 8 waves
  const int wm   = w >> 2;     // 0..1  -> 64 rows of M each
  const int wn   = w & 3;      // 0..3  -> 32 cols of N each
  const int m0   = blockIdx.y * TM;
  const int n0   = blockIdx.x * TN;

  v8f acc[4][2];
  const v8f vzero = {0.f, 0.f, 0.f, 0.f, 0.f, 0.f, 0.f, 0.f};
#pragma unroll
  for (int i = 0; i < 4; ++i)
#pragma unroll
    for (int j = 0; j < 2; ++j) acc[i][j] = vzero;

  const int mr = lane & 15;          // row index inside 16x16 tile
  const int hi = (lane >> 4) << 4;   // byte offset selecting K 0..7 vs 8..15 halves

  // stage one 128x64 A-tile + 128x64 W-tile into LDS buffer `bufi` (8 x 16B per thread)
  auto stage = [&](int bufi, int kk) {
#pragma unroll
    for (int rep = 0; rep < 4; ++rep) {
      int idx = t + rep * 256;       // 0..1023
      int row = idx >> 3;            // 0..127
      int seg = idx & 7;             // 16B segment within 128B row
      const bf16* ga = A + (size_t)(m0 + row) * K + kk + seg * 8;
      const bf16* gb = W + (size_t)(n0 + row) * K + kk + seg * 8;
      unsigned char* la = &sA[bufi][row * PITCH + seg * 16];
      unsigned char* lb = &sB[bufi][row * PITCH + seg * 16];
#if HAVE_ASYNC_LDS
      __builtin_amdgcn_global_load_async_to_lds_b128(to_global_v4(ga), to_lds_v4(la), 0, 0);
      __builtin_amdgcn_global_load_async_to_lds_b128(to_global_v4(gb), to_lds_v4(lb), 0, 0);
#else
      U4 ra = *(const U4*)ga;
      U4 rb = *(const U4*)gb;
      *(U4*)la = ra;
      *(U4*)lb = rb;
#endif
    }
  };

  stage(0, 0);                       // prologue: fill buffer 0
  int cb = 0;

  for (int k0 = 0; k0 < K; k0 += TK) {
    const bool has_next = (k0 + TK) < K;
    if (has_next) {
      stage(cb ^ 1, k0 + TK);        // overlap next tile's DMA with this tile's WMMA
      wait_async_le8();              // in-order completion -> current tile has landed
    } else {
      wait_async_le0();
    }
    __syncthreads();

    // ---- 2 K-substeps of 32, 8 WMMAs each (16 per barrier window)
#pragma unroll
    for (int s = 0; s < 2; ++s) {
      const int so = s * 64;         // byte offset of K-substep within the 128B row
      Frag a[4], b[2];
#pragma unroll
      for (int i = 0; i < 4; ++i) {
        const unsigned char* p = &sA[cb][(wm * 64 + i * 16 + mr) * PITCH + so];
        a[i].q[0] = *(const U4*)(p + hi);        // K 0..7  / 8..15
        a[i].q[1] = *(const U4*)(p + 32 + hi);   // K 16..23 / 24..31
      }
#pragma unroll
      for (int j = 0; j < 2; ++j) {
        const unsigned char* p = &sB[cb][(wn * 32 + j * 16 + mr) * PITCH + so];
        b[j].q[0] = *(const U4*)(p + hi);
        b[j].q[1] = *(const U4*)(p + 32 + hi);
      }
#pragma unroll
      for (int i = 0; i < 4; ++i)
#pragma unroll
        for (int j = 0; j < 2; ++j)
          acc[i][j] = __builtin_amdgcn_wmma_f32_16x16x32_bf16(
              false, a[i].v, false, b[j].v, (short)0, acc[i][j], false, false);
    }

    __syncthreads();                 // all waves done reading cb before it is refilled
    cb ^= 1;
  }

  // ---- store: C/D layout -> lane n = lane&15, m = jj + 8*(lane>>4)
  const int nc = n0 + wn * 32 + mr;
  const int mhi = (lane >> 4) * 8;
#pragma unroll
  for (int j = 0; j < 2; ++j) {
    int n = nc + j * 16;
    float bv = bias[n];
#pragma unroll
    for (int i = 0; i < 4; ++i) {
#pragma unroll
      for (int jj = 0; jj < 8; ++jj) {
        int m = m0 + wm * 64 + i * 16 + mhi + jj;
        float v = acc[i][j][jj] + bv;
        if constexpr (OUT_F32) outF[(size_t)m * N + n] = v;
        else                   outB[(size_t)m * N + n] = (bf16)v;
      }
    }
  }
}

// ---------------------------------------------------------------- scan phase 1: per-chunk local recurrence
__global__ __launch_bounds__(256) void scan_local(const bf16* __restrict__ u,
                                                  bf16* __restrict__ x,
                                                  float* __restrict__ finals,
                                                  const float* __restrict__ params_log) {
  int gid = blockIdx.x * 256 + threadIdx.x;       // b*NCHUNK*U + c*U + uidx
  int uidx = gid & (U_ - 1);
  int c    = (gid >> 10) & (NCHUNK - 1);
  int b    = gid >> 16;
  float nu  = __expf(params_log[uidx]);           // params_log[0][u]
  float lam = __expf(-nu);
  size_t base = ((size_t)b * L_ + (size_t)c * CHUNK) * U_ + uidx;
  float acc = 0.f;
#pragma unroll 4
  for (int tt = 0; tt < CHUNK; ++tt) {
    float v = (float)u[base + (size_t)tt * U_];
    acc = fmaf(acc, lam, v);
    x[base + (size_t)tt * U_] = (bf16)acc;
  }
  finals[gid] = acc;
}

// ---------------------------------------------------------------- scan phase 2: carry across chunks
__global__ __launch_bounds__(256) void scan_carry(const float* __restrict__ finals,
                                                  float* __restrict__ carries,
                                                  const float* __restrict__ params_log) {
  int gid = blockIdx.x * 256 + threadIdx.x;       // b*U + uidx
  int uidx = gid & (U_ - 1);
  int b    = gid >> 10;
  float nu = __expf(params_log[uidx]);
  float lamC = __expf(-nu * (float)CHUNK);
  float e = 0.f;
  for (int c = 0; c < NCHUNK; ++c) {
    size_t idx = ((size_t)(b * NCHUNK + c)) * U_ + uidx;
    carries[idx] = e;                             // carry INTO chunk c
    e = fmaf(e, lamC, finals[idx]);
  }
}

// ---------------------------------------------------------------- scan phase 3: apply carry + gamma
__global__ __launch_bounds__(256) void scan_fix(bf16* __restrict__ x,
                                                const float* __restrict__ carries,
                                                const float* __restrict__ params_log) {
  int gid = blockIdx.x * 256 + threadIdx.x;
  int uidx = gid & (U_ - 1);
  int c    = (gid >> 10) & (NCHUNK - 1);
  int b    = gid >> 16;
  float nu    = __expf(params_log[uidx]);
  float gamma = __expf(params_log[U_ + uidx]);    // params_log[1][u]
  float lam   = __expf(-nu);
  float carry = carries[gid];
  size_t base = ((size_t)b * L_ + (size_t)c * CHUNK) * U_ + uidx;
  float p = lam;                                  // lam^(t_local+1)
#pragma unroll 4
  for (int tt = 0; tt < CHUNK; ++tt) {
    float v = (float)x[base + (size_t)tt * U_] + p * carry;
    x[base + (size_t)tt * U_] = (bf16)(v * gamma);
    p *= lam;
  }
}

// ---------------------------------------------------------------- launch
extern "C" void kernel_launch(void* const* d_in, const int* in_sizes, int n_in,
                              void* d_out, int out_size, void* d_ws, size_t ws_size,
                              hipStream_t stream) {
  const float* inp        = (const float*)d_in[0];  // (B,L,H)
  const float* Wi         = (const float*)d_in[1];  // (U,H)
  const float* bi         = (const float*)d_in[2];  // (U,)
  const float* Wo         = (const float*)d_in[3];  // (H,U)
  const float* bo         = (const float*)d_in[4];  // (H,)
  const float* params_log = (const float*)d_in[5];  // (2,U)
  float* out = (float*)d_out;

  char* w = (char*)d_ws;
  bf16* inB = (bf16*)w;  w += (size_t)BL_ * H_ * 2;      // 33.5 MB
  bf16* WiB = (bf16*)w;  w += (size_t)U_ * H_ * 2;       //  2 MB
  bf16* WoB = (bf16*)w;  w += (size_t)H_ * U_ * 2;       //  2 MB
  bf16* uB  = (bf16*)w;  w += (size_t)BL_ * U_ * 2;      // 33.5 MB
  bf16* xB  = (bf16*)w;  w += (size_t)BL_ * U_ * 2;      // 33.5 MB
  float* finals  = (float*)w; w += (size_t)B_ * NCHUNK * U_ * 4;  // 1 MB
  float* carries = (float*)w;                                     // 1 MB

  // 1) f32 -> bf16 conversions
  cvt_f32_bf16<<<(BL_ * H_ + 255) / 256, 256, 0, stream>>>(inp, inB, BL_ * H_);
  cvt_f32_bf16<<<(U_ * H_ + 255) / 256, 256, 0, stream>>>(Wi, WiB, U_ * H_);
  cvt_f32_bf16<<<(H_ * U_ + 255) / 256, 256, 0, stream>>>(Wo, WoB, H_ * U_);

  // 2) u = inputs @ Wi^T + bi   (bf16 out)
  dim3 g1(U_ / TN, BL_ / TM);
  gemm_bf16_wmma<false><<<g1, 256, 0, stream>>>(inB, WiB, bi, nullptr, uB,
                                                BL_, U_, H_);

  // 3) linear recurrence (chunked scan) + gamma
  scan_local<<<(B_ * NCHUNK * U_) / 256, 256, 0, stream>>>(uB, xB, finals, params_log);
  scan_carry<<<(B_ * U_) / 256, 256, 0, stream>>>(finals, carries, params_log);
  scan_fix<<<(B_ * NCHUNK * U_) / 256, 256, 0, stream>>>(xB, carries, params_log);

  // 4) out = x @ Wo^T + bo   (f32 out)
  dim3 g2(H_ / TN, BL_ / TM);
  gemm_bf16_wmma<true><<<g2, 256, 0, stream>>>(xB, WoB, bo, out, nullptr,
                                               BL_, H_, U_);
}
